// MultiViewCorrelationModule_61186104099403
// MI455X (gfx1250) — matrix operand order repaired and checked by reading, not compiled, not run on hardware
//
#include <hip/hip_runtime.h>

typedef __attribute__((ext_vector_type(16))) _Float16 v16h;
typedef __attribute__((ext_vector_type(8)))  _Float16 v8h;
typedef __attribute__((ext_vector_type(8)))  float    v8f;
typedef __attribute__((ext_vector_type(4)))  float    v4f;
typedef __attribute__((ext_vector_type(2)))  float    v2f;

constexpr int Bn    = 4096;   // batch
constexpr int Mrows = 16;     // items per outfit
constexpr int Dd    = 1024;   // feature dim per item
constexpr int FLAT  = 16384;  // M*D
constexpr int H1    = 512;    // hidden
constexpr int OUTF  = 128;    // output features

// ---------------------------------------------------------------- utilities

__device__ inline float reduce_block(float v, float* red, int n) {
  const int tid = threadIdx.x;
  red[tid] = v;
  __syncthreads();
  for (int w = n >> 1; w > 0; w >>= 1) {
    if (tid < w) red[tid] += red[tid + w];
    __syncthreads();
  }
  float r = red[0];
  __syncthreads();
  return r;
}

// 128x128 f32 matmul using v_wmma_f32_16x16x4_f32; callers are 256-thread
// single blocks; caller provides barriers between dependent calls.
__device__ inline void mm128(const float* __restrict__ A, const float* __restrict__ Bm,
                             float* __restrict__ C, float scale, float diag) {
  const int wave = threadIdx.x >> 5;
  const int lane = threadIdx.x & 31;
  const int half = (lane >> 4) & 1;
  const int r    = lane & 15;
  for (int t = wave; t < 64; t += 8) {
    const int ti = t >> 3, tj = t & 7;
    v8f acc = {};
    for (int k0 = 0; k0 < 128; k0 += 4) {
      const int k = k0 + half * 2;
      v2f a, b;
      a[0] = A[(ti * 16 + r) * 128 + k];
      a[1] = A[(ti * 16 + r) * 128 + k + 1];
      b[0] = Bm[(size_t)k * 128 + tj * 16 + r];
      b[1] = Bm[(size_t)(k + 1) * 128 + tj * 16 + r];
      acc = __builtin_amdgcn_wmma_f32_16x16x4_f32(false, a, false, b, (short)0, acc,
                                                  false, false);
    }
#pragma unroll
    for (int i = 0; i < 8; ++i) {
      const int row = ti * 16 + i + 8 * half;
      const int col = tj * 16 + r;
      C[row * 128 + col] = acc[i] * scale + ((row == col) ? diag : 0.f);
    }
  }
}

// ---------------------------------------------------------------- kernels

// per-sample LayerNorm stats over masked x (shared by both towers, since G is
// a per-sample permutation of H) + valid-item counts
__global__ __launch_bounds__(256)
void stats_kernel(const float* __restrict__ x, const int* __restrict__ mask,
                  float* __restrict__ mu, float* __restrict__ rstd,
                  int* __restrict__ counts) {
  __shared__ float red[256];
  __shared__ int smask[16];
  const int b = blockIdx.x, tid = threadIdx.x;
  if (tid < 16) smask[tid] = mask[b * Mrows + tid];
  __syncthreads();
  float s = 0.f, s2 = 0.f;
  const float* xb = x + (size_t)b * FLAT;
  for (int e = tid; e < FLAT; e += 256) {
    float v = smask[e >> 10] ? xb[e] : 0.f;
    s += v;
    s2 += v * v;
  }
  float tot  = reduce_block(s, red, 256);
  float tot2 = reduce_block(s2, red, 256);
  if (tid == 0) {
    float mean = tot * (1.f / FLAT);
    float var  = tot2 * (1.f / FLAT) - mean * mean;
    mu[b]   = mean;
    rstd[b] = rsqrtf(var + 1e-5f);
    int c = 0;
    for (int m = 0; m < Mrows; ++m) c += smask[m] ? 1 : 0;
    counts[b] = c;
  }
}

__global__ __launch_bounds__(256)
void cast_kernel(const float* __restrict__ src, _Float16* __restrict__ dst, int n) {
  const int stride = gridDim.x * blockDim.x;
  for (int i = blockIdx.x * blockDim.x + threadIdx.x; i < n; i += stride)
    dst[i] = (_Float16)src[i];
}

// Fused (mask/reverse-gather + LayerNorm affine) -> f16 LDS tile -> WMMA GEMM
// vs W^T, bias + ReLU epilogue -> f16 mid activations.  towerG selects the
// reversed (G) gather; LN stats are shared between towers.
// One block per 16-sample group; each of 8 waves owns 4 N-tiles (full N=512),
// so the A tile (and x) is read from HBM exactly once.
__global__ __launch_bounds__(256)
void gemm1_kernel(const float* __restrict__ x, const _Float16* __restrict__ W16,
                  const float* __restrict__ g, const float* __restrict__ beta,
                  const float* __restrict__ bias, const float* __restrict__ mu,
                  const float* __restrict__ rstd, const int* __restrict__ mask,
                  const int* __restrict__ counts, _Float16* __restrict__ Mid16,
                  int towerG) {
  __shared__ __align__(16) _Float16 As[16][136];
  __shared__ int   esrc[16][16];
  __shared__ float smu[16], srstd[16];
  const int tid  = threadIdx.x;
  const int b0   = blockIdx.x * 16;
  const int wave = tid >> 5, lane = tid & 31;
  const int half = (lane >> 4) & 1, r = lane & 15;

  {
    const int row = tid >> 4, m = tid & 15;
    int s;
    if (towerG) {
      const int src = counts[b0 + row] - 1 - m;
      s = (src >= 0) ? src : -1;
    } else {
      s = mask[(b0 + row) * Mrows + m] ? m : -1;
    }
    esrc[row][m] = s;
    if (tid < 16) { smu[tid] = mu[b0 + tid]; srstd[tid] = rstd[b0 + tid]; }
  }
  __syncthreads();

  const int frow = tid >> 4;        // fill: row handled by this thread
  const int fk0  = (tid & 15) * 8;  // fill: 8 contiguous cols within chunk

  v8f acc[4] = {{}, {}, {}, {}};
  for (int kc0 = 0; kc0 < FLAT; kc0 += 128) {
    // branchless vectorized fill: within a 128-wide chunk, m = k>>10 is
    // uniform, so the gather source row is a per-row scalar.
    {
      const int mchunk = kc0 >> 10;
      const int s      = esrc[frow][mchunk];
      const float vf   = (s >= 0) ? 1.f : 0.f;
      const int  srow  = (s >= 0) ? s : 0;
      const float* xp = x + ((size_t)(b0 + frow) * Mrows + srow) * Dd +
                        (kc0 & 1023) + fk0;
      const float* gp = g + kc0 + fk0;
      const float* bp = beta + kc0 + fk0;
      const float muv = smu[frow], rs = srstd[frow];
      v4f xa = ((const v4f*)xp)[0], xb = ((const v4f*)xp)[1];
      v4f ga = ((const v4f*)gp)[0], gb = ((const v4f*)gp)[1];
      v4f ba = ((const v4f*)bp)[0], bb = ((const v4f*)bp)[1];
      v8h hv;
#pragma unroll
      for (int i = 0; i < 4; ++i) {
        hv[i]     = (_Float16)((xa[i] * vf - muv) * rs * ga[i] + ba[i]);
        hv[4 + i] = (_Float16)((xb[i] * vf - muv) * rs * gb[i] + bb[i]);
      }
      *(v8h*)&As[frow][fk0] = hv;
    }
    __syncthreads();
#pragma unroll
    for (int kb = 0; kb < 4; ++kb) {
      const int k0 = kb * 32;
      const _Float16* arow = &As[r][k0 + half * 16];
      v8h alo = *(const v8h*)(arow);
      v8h ahi = *(const v8h*)(arow + 8);
      v16h av = __builtin_shufflevector(alo, ahi, 0, 1, 2, 3, 4, 5, 6, 7, 8, 9, 10,
                                        11, 12, 13, 14, 15);
#pragma unroll
      for (int p = 0; p < 4; ++p) {
        const int ncol = wave * 16 + p * 128 + r;
        const _Float16* wb = W16 + (size_t)ncol * FLAT + kc0 + k0 + half * 16;
        v16h bv = *(const v16h*)wb;
        acc[p] = __builtin_amdgcn_wmma_f32_16x16x32_f16(false, av, false, bv,
                                                        (short)0, acc[p], false,
                                                        false);
      }
    }
    __syncthreads();
  }
#pragma unroll
  for (int p = 0; p < 4; ++p) {
    const int col = wave * 16 + p * 128 + r;
    const float bb = bias[col];
#pragma unroll
    for (int i = 0; i < 8; ++i) {
      const int row = b0 + i + 8 * half;
      float val = fmaxf(acc[p][i] + bb, 0.f);
      Mid16[(size_t)row * H1 + col] = (_Float16)val;
    }
  }
}

// [B x 512] @ [512 x 128] (w1b^T), bias epilogue, f32 out
__global__ __launch_bounds__(256)
void gemm2_kernel(const _Float16* __restrict__ Mid16, const _Float16* __restrict__ Wb16,
                  const float* __restrict__ bias, float* __restrict__ F) {
  const int tid  = threadIdx.x;
  const int b0   = blockIdx.x * 16;
  const int wave = tid >> 5, lane = tid & 31;
  const int half = (lane >> 4) & 1, r = lane & 15;
  const int col0 = wave * 16;
  v8f acc = {};
  const _Float16* arow = Mid16 + (size_t)(b0 + r) * H1;
  const _Float16* brow = Wb16 + (size_t)(col0 + r) * H1 + half * 16;
#pragma unroll 4
  for (int k0 = 0; k0 < H1; k0 += 32) {
    v8h alo = *(const v8h*)(arow + k0 + half * 16);
    v8h ahi = *(const v8h*)(arow + k0 + half * 16 + 8);
    v16h av = __builtin_shufflevector(alo, ahi, 0, 1, 2, 3, 4, 5, 6, 7, 8, 9, 10, 11,
                                      12, 13, 14, 15);
    v16h bv = *(const v16h*)(brow + k0);
    acc = __builtin_amdgcn_wmma_f32_16x16x32_f16(false, av, false, bv, (short)0, acc,
                                                 false, false);
  }
  const int col = col0 + r;
  const float bb = bias[col];
#pragma unroll
  for (int i = 0; i < 8; ++i)
    F[(size_t)(b0 + i + 8 * half) * OUTF + col] = acc[i] + bb;
}

// column means of F_H / F_G over the batch
__global__ __launch_bounds__(256)
void colmean_kernel(const float* __restrict__ FH, const float* __restrict__ FG,
                    float* __restrict__ meanH, float* __restrict__ meanG) {
  __shared__ float red[256];
  const int j   = blockIdx.x;
  const float* F = (j < OUTF) ? FH : FG;
  const int col = j & (OUTF - 1);
  float s = 0.f;
  for (int b = threadIdx.x; b < Bn; b += 256) s += F[(size_t)b * OUTF + col];
  float tot = reduce_block(s, red, 256);
  if (threadIdx.x == 0) {
    float m = tot / Bn;
    if (j < OUTF) meanH[col] = m; else meanG[col] = m;
  }
}

// out = n1 * (FA - meanA)^T (FB - meanB) + diag*I  via f32 WMMA, K = 4096
__global__ __launch_bounds__(32)
void cov_kernel(const float* __restrict__ FA, const float* __restrict__ meanA,
                const float* __restrict__ FB, const float* __restrict__ meanB,
                float* __restrict__ out, float diag, float n1) {
  const int lane = threadIdx.x & 31;
  const int half = (lane >> 4) & 1, r = lane & 15;
  const int bi = blockIdx.x, bj = blockIdx.y;
  const float am = meanA[bi * 16 + r];
  const float bm = meanB[bj * 16 + r];
  v8f acc = {};
  for (int k0 = 0; k0 < Bn; k0 += 4) {
    const int k = k0 + half * 2;
    v2f a, b;
    a[0] = FA[(size_t)k * OUTF + bi * 16 + r] - am;
    a[1] = FA[(size_t)(k + 1) * OUTF + bi * 16 + r] - am;
    b[0] = FB[(size_t)k * OUTF + bj * 16 + r] - bm;
    b[1] = FB[(size_t)(k + 1) * OUTF + bj * 16 + r] - bm;
    acc = __builtin_amdgcn_wmma_f32_16x16x4_f32(false, a, false, b, (short)0, acc,
                                                false, false);
  }
#pragma unroll
  for (int i = 0; i < 8; ++i) {
    const int row = bi * 16 + i + 8 * half;
    const int col = bj * 16 + r;
    out[row * OUTF + col] = acc[i] * n1 + ((row == col) ? diag : 0.f);
  }
}

// Newton-Schulz inverse square root of a 128x128 SPD matrix (single block)
__global__ __launch_bounds__(256)
void invsqrt128_kernel(const float* __restrict__ S, float* __restrict__ out,
                       float* Y, float* Y2, float* Z, float* Z2, float* Tm) {
  __shared__ float red[256];
  const int tid = threadIdx.x;
  float s = 0.f;
  for (int e = tid; e < OUTF * OUTF; e += 256) { float v = S[e]; s += v * v; }
  float fro2 = reduce_block(s, red, 256);
  float c = sqrtf(fro2) + 1e-12f;
  float ic = 1.f / c;
  for (int e = tid; e < OUTF * OUTF; e += 256) {
    Y[e] = S[e] * ic;
    Z[e] = ((e >> 7) == (e & 127)) ? 1.f : 0.f;
  }
  __syncthreads();
  float *y = Y, *y2 = Y2, *z = Z, *z2 = Z2;
  for (int it = 0; it < 14; ++it) {
    mm128(z, y, Tm, -0.5f, 1.5f);   // Tm = 1.5 I - 0.5 Z Y
    __syncthreads();
    mm128(y, Tm, y2, 1.f, 0.f);     // Y' = Y Tm
    mm128(Tm, z, z2, 1.f, 0.f);     // Z' = Tm Z
    __syncthreads();
    float* t;
    t = y; y = y2; y2 = t;
    t = z; z = z2; z2 = t;
  }
  float sc = rsqrtf(c);
  for (int e = tid; e < OUTF * OUTF; e += 256) out[e] = z[e] * sc;
}

// T = ISHH @ SHG @ ISGG  (written straight into d_out)
__global__ __launch_bounds__(256)
void twomm_kernel(const float* __restrict__ ISHH, const float* __restrict__ SHG,
                  const float* __restrict__ ISGG, float* __restrict__ TMP,
                  float* __restrict__ Tout) {
  mm128(ISHH, SHG, TMP, 1.f, 0.f);
  __syncthreads();
  mm128(TMP, ISGG, Tout, 1.f, 0.f);
}

// top singular values of T via A = T^T T, orthogonal iteration (p=16) +
// 16x16 Jacobi Rayleigh-Ritz; writes sigmoid(mean top-10 sigma) to scores
__global__ __launch_bounds__(128)
void sv_kernel(const float* __restrict__ Tmat, float* __restrict__ Tt,
               float* __restrict__ Amat, float* __restrict__ Q,
               float* __restrict__ Zq, float* __restrict__ Wq,
               float* __restrict__ scores) {
  __shared__ float red[128];
  __shared__ float sval;
  const int tid = threadIdx.x;
  for (int e = tid; e < OUTF * OUTF; e += 128)
    Tt[e] = Tmat[(e & 127) * OUTF + (e >> 7)];
  __syncthreads();
  for (int e = tid; e < OUTF * OUTF; e += 128) {
    const int i = e >> 7, j = e & 127;
    float s = 0.f;
    for (int k = 0; k < OUTF; ++k) s += Tt[i * OUTF + k] * Tmat[k * OUTF + j];
    Amat[e] = s;
  }
  __syncthreads();
  for (int e = tid; e < OUTF * 16; e += 128) {
    const int i = e >> 4, j = e & 15;
    Q[e] = (i == j) ? 1.f : 0.f;
  }
  __syncthreads();
  for (int it = 0; it < 24; ++it) {
    for (int e = tid; e < OUTF * 16; e += 128) {
      const int i = e >> 4, j = e & 15;
      float s = 0.f;
      for (int k = 0; k < OUTF; ++k) s += Amat[i * OUTF + k] * Q[k * 16 + j];
      Zq[e] = s;
    }
    __syncthreads();
    for (int j = 0; j < 16; ++j) {            // modified Gram-Schmidt
      float v = Zq[tid * 16 + j];
      float n2 = reduce_block(v * v, red, 128);
      float inv = rsqrtf(n2 + 1e-20f);
      Q[tid * 16 + j] = Zq[tid * 16 + j] * inv;
      __syncthreads();
      for (int k2 = j + 1; k2 < 16; ++k2) {
        float d = reduce_block(Q[tid * 16 + j] * Zq[tid * 16 + k2], red, 128);
        Zq[tid * 16 + k2] -= d * Q[tid * 16 + j];
        __syncthreads();
      }
    }
  }
  for (int e = tid; e < OUTF * 16; e += 128) {   // Wq = A Q
    const int i = e >> 4, j = e & 15;
    float s = 0.f;
    for (int k = 0; k < OUTF; ++k) s += Amat[i * OUTF + k] * Q[k * 16 + j];
    Wq[e] = s;
  }
  __syncthreads();
  for (int e = tid; e < 256; e += 128) {         // R = Q^T Wq (16x16) in Zq
    const int i = e >> 4, j = e & 15;
    float s = 0.f;
    for (int k = 0; k < OUTF; ++k) s += Q[k * 16 + i] * Wq[k * 16 + j];
    Zq[e] = s;
  }
  __syncthreads();
  if (tid == 0) {
    float a[16][16];
    for (int i = 0; i < 16; ++i)
      for (int j = 0; j < 16; ++j) a[i][j] = Zq[i * 16 + j];
    for (int sweep = 0; sweep < 20; ++sweep)
      for (int p = 0; p < 15; ++p)
        for (int q = p + 1; q < 16; ++q) {
          float apq = a[p][q];
          if (fabsf(apq) < 1e-12f) continue;
          float tau = (a[q][q] - a[p][p]) / (2.f * apq);
          float t = ((tau >= 0.f) ? 1.f : -1.f) / (fabsf(tau) + sqrtf(1.f + tau * tau));
          float cc = rsqrtf(1.f + t * t), ss = t * cc;
          for (int k = 0; k < 16; ++k) {
            float akp = a[k][p], akq = a[k][q];
            a[k][p] = cc * akp - ss * akq;
            a[k][q] = ss * akp + cc * akq;
          }
          for (int k = 0; k < 16; ++k) {
            float apk = a[p][k], aqk = a[q][k];
            a[p][k] = cc * apk - ss * aqk;
            a[q][k] = ss * apk + cc * aqk;
          }
        }
    float ev[16];
    for (int i = 0; i < 16; ++i) ev[i] = a[i][i];
    for (int i = 0; i < 16; ++i)
      for (int j = i + 1; j < 16; ++j)
        if (ev[j] > ev[i]) { float t = ev[i]; ev[i] = ev[j]; ev[j] = t; }
    float qsum = 0.f;
    for (int i = 0; i < 10; ++i) qsum += sqrtf(fmaxf(ev[i], 0.f));
    sval = 1.f / (1.f + expf(-qsum * 0.1f));
  }
  __syncthreads();
  const float val = sval;
  for (int e = tid; e < Bn; e += 128) scores[e] = val;
}

// ---------------------------------------------------------------- launch

extern "C" void kernel_launch(void* const* d_in, const int* in_sizes, int n_in,
                              void* d_out, int out_size, void* d_ws, size_t ws_size,
                              hipStream_t stream) {
  const float* x     = (const float*)d_in[0];
  const int*   maskp = (const int*)  d_in[1];
  const float* ln1g  = (const float*)d_in[2];
  const float* ln1b  = (const float*)d_in[3];
  const float* w1a   = (const float*)d_in[4];
  const float* b1a   = (const float*)d_in[5];
  const float* w1b   = (const float*)d_in[6];
  const float* b1b   = (const float*)d_in[7];
  const float* ln2g  = (const float*)d_in[8];
  const float* ln2b  = (const float*)d_in[9];
  const float* w2a   = (const float*)d_in[10];
  const float* b2a   = (const float*)d_in[11];
  const float* w2b   = (const float*)d_in[12];
  const float* b2b   = (const float*)d_in[13];

  char* ws = (char*)d_ws;
  size_t off = 0;
  auto take = [&](size_t bytes) -> char* {
    char* p = ws + off;
    off += (bytes + 255) & ~(size_t)255;
    return p;
  };
  float* mu       = (float*)take((size_t)Bn * 4);
  float* rstd     = (float*)take((size_t)Bn * 4);
  int*   cnts     = (int*)  take((size_t)Bn * 4);
  _Float16* W1a16 = (_Float16*)take((size_t)H1 * FLAT * 2);
  _Float16* W2a16 = (_Float16*)take((size_t)H1 * FLAT * 2);
  _Float16* W1b16 = (_Float16*)take((size_t)OUTF * H1 * 2);
  _Float16* W2b16 = (_Float16*)take((size_t)OUTF * H1 * 2);
  _Float16* Hmid  = (_Float16*)take((size_t)Bn * H1 * 2);
  _Float16* Gmid  = (_Float16*)take((size_t)Bn * H1 * 2);
  float* FH    = (float*)take((size_t)Bn * OUTF * 4);
  float* FG    = (float*)take((size_t)Bn * OUTF * 4);
  float* meanH = (float*)take(OUTF * 4);
  float* meanG = (float*)take(OUTF * 4);
  float* SHH   = (float*)take(OUTF * OUTF * 4);
  float* SGG   = (float*)take(OUTF * OUTF * 4);
  float* SHG   = (float*)take(OUTF * OUTF * 4);
  float* ISHH  = (float*)take(OUTF * OUTF * 4);
  float* ISGG  = (float*)take(OUTF * OUTF * 4);
  float* Ybuf  = (float*)take(OUTF * OUTF * 4);
  float* Y2buf = (float*)take(OUTF * OUTF * 4);
  float* Zbuf  = (float*)take(OUTF * OUTF * 4);
  float* Z2buf = (float*)take(OUTF * OUTF * 4);
  float* Tmbuf = (float*)take(OUTF * OUTF * 4);
  float* TMP   = (float*)take(OUTF * OUTF * 4);
  float* Tt    = (float*)take(OUTF * OUTF * 4);
  float* Amat  = (float*)take(OUTF * OUTF * 4);
  float* Qm    = (float*)take(OUTF * 16 * 4);
  float* Zq    = (float*)take(OUTF * 16 * 4);
  float* Wq    = (float*)take(OUTF * 16 * 4);

  float* scores = (float*)d_out;        // [B] (B x 1)
  float* Tout   = (float*)d_out + Bn;   // [128 x 128]

  stats_kernel<<<Bn, 256, 0, stream>>>(x, maskp, mu, rstd, cnts);
  cast_kernel<<<2048, 256, 0, stream>>>(w1a, W1a16, H1 * FLAT);
  cast_kernel<<<2048, 256, 0, stream>>>(w2a, W2a16, H1 * FLAT);
  cast_kernel<<<64, 256, 0, stream>>>(w1b, W1b16, OUTF * H1);
  cast_kernel<<<64, 256, 0, stream>>>(w2b, W2b16, OUTF * H1);

  gemm1_kernel<<<Bn / 16, 256, 0, stream>>>(
      x, W1a16, ln1g, ln1b, b1a, mu, rstd, maskp, cnts, Hmid, 0);
  gemm1_kernel<<<Bn / 16, 256, 0, stream>>>(
      x, W2a16, ln2g, ln2b, b2a, mu, rstd, maskp, cnts, Gmid, 1);

  gemm2_kernel<<<Bn / 16, 256, 0, stream>>>(Hmid, W1b16, b1b, FH);
  gemm2_kernel<<<Bn / 16, 256, 0, stream>>>(Gmid, W2b16, b2b, FG);

  colmean_kernel<<<2 * OUTF, 256, 0, stream>>>(FH, FG, meanH, meanG);

  const float n1 = 1.0f / (Bn - 1);
  cov_kernel<<<dim3(8, 8), 32, 0, stream>>>(FH, meanH, FH, meanH, SHH, 1e-4f, n1);
  cov_kernel<<<dim3(8, 8), 32, 0, stream>>>(FG, meanG, FG, meanG, SGG, 1e-4f, n1);
  cov_kernel<<<dim3(8, 8), 32, 0, stream>>>(FH, meanH, FG, meanG, SHG, 0.f, n1);

  invsqrt128_kernel<<<1, 256, 0, stream>>>(SHH, ISHH, Ybuf, Y2buf, Zbuf, Z2buf, Tmbuf);
  invsqrt128_kernel<<<1, 256, 0, stream>>>(SGG, ISGG, Ybuf, Y2buf, Zbuf, Z2buf, Tmbuf);

  twomm_kernel<<<1, 256, 0, stream>>>(ISHH, SHG, ISGG, TMP, Tout);
  sv_kernel<<<1, 128, 0, stream>>>(Tout, Tt, Amat, Qm, Zq, Wq, scores);

  (void)in_sizes; (void)n_in; (void)out_size; (void)ws_size;
}
